// DiffusionBlock_68968584839220
// MI455X (gfx1250) — matrix operand order repaired and checked by reading, not compiled, not run on hardware
//
#include <hip/hip_runtime.h>
#include <math.h>

#define BB 2
#define NN 2048
#define DD 1024
#define HH 16
#define HCC 8
#define HID 4096
#define DH 64
#define DHC 128

typedef __attribute__((ext_vector_type(16))) __bf16 v16bf;
typedef __attribute__((ext_vector_type(8)))  __bf16 v8bf;
typedef __attribute__((ext_vector_type(8)))  float  v8f;

union U16 { v16bf v; v8bf h[2]; unsigned short s[16]; };

__device__ __forceinline__ unsigned short f2bf(float f) {
    union { float f; unsigned u; } a; a.f = f;
    unsigned r = a.u + 0x7FFFu + ((a.u >> 16) & 1u);
    return (unsigned short)(r >> 16);
}
__device__ __forceinline__ float bf2f(unsigned short s) {
    union { unsigned u; float f; } a; a.u = ((unsigned)s) << 16;
    return a.f;
}
__device__ __forceinline__ v8f v8f_zero() {
    v8f z;
#pragma unroll
    for (int i = 0; i < 8; i++) z[i] = 0.f;
    return z;
}
__device__ __forceinline__ v8f wmma_bf16(v16bf a, v16bf b, v8f c) {
    return __builtin_amdgcn_wmma_f32_16x16x32_bf16(false, a, false, b, (short)0, c, false, false);
}

// A fragment (16x32 bf16 tile, row-major, `stride` halves per row).
// lanes 0-15: M=lane, K = kh..kh+15, kh = (lane/16)*16
__device__ __forceinline__ v16bf frag_a(const unsigned short* tile, int stride, int lane) {
    int row = lane & 15;
    int kh  = (lane >> 4) << 4;
    U16 u;
    u.h[0] = *(const v8bf*)(tile + (size_t)row * stride + kh);
    u.h[1] = *(const v8bf*)(tile + (size_t)row * stride + kh + 8);
    return u.v;
}
// B fragment where this lane's 16 K-values are contiguous at p (2 x b128 loads).
// lanes 0-15: K=0..15 col=lane; lanes 16-31: K=16..31 col=lane-16.
__device__ __forceinline__ v16bf frag_b_rows(const unsigned short* p) {
    U16 u;
    u.h[0] = *(const v8bf*)(p);
    u.h[1] = *(const v8bf*)(p + 8);
    return u.v;
}

// ---------------- timestep embedding ----------------
__global__ void k_embed(const float* __restrict__ t, float* __restrict__ emb) {
    int idx = blockIdx.x * 256 + threadIdx.x;   // B*HID = 8192
    int b = idx >> 12;
    int i = idx & 4095;
    int j = i & 2047;
    float f = expf(-logf(10000.0f) * (float)j / 2048.0f);
    float arg = t[b] * f;
    emb[idx] = (i < 2048) ? cosf(arg) : sinf(arg);
}

// ---------------- skinny GEMV (B=2), memory bound ----------------
__global__ void k_gemv(const float* __restrict__ in, const float* __restrict__ W,
                       const float* __restrict__ bias, float* __restrict__ out,
                       int K, int fout, int act) {
    int o = blockIdx.x * 256 + threadIdx.x;
    float a0 = 0.f, a1 = 0.f;
#pragma unroll 4
    for (int k = 0; k < K; k++) {
        float w = W[(size_t)k * fout + o];
        a0 += in[k] * w;
        a1 += in[K + k] * w;
    }
    a0 += bias[o]; a1 += bias[o];
    if (act) {
        a0 = a0 / (1.f + expf(-a0));
        a1 = a1 / (1.f + expf(-a1));
    }
    out[o] = a0;
    out[fout + o] = a1;
}

// ---------------- f32 (K,N) -> bf16 transposed (N,K) ----------------
__global__ void k_f2bf_t(const float* __restrict__ src, unsigned short* __restrict__ dst,
                         int K, int Nout) {
    __shared__ unsigned short tile[32][33];
    int k0 = blockIdx.x * 32, n0 = blockIdx.y * 32;
    int tx = threadIdx.x, ty = threadIdx.y;   // (32,8)
#pragma unroll
    for (int j = 0; j < 4; j++) {
        int k = ty + j * 8;
        tile[k][tx] = f2bf(src[(size_t)(k0 + k) * Nout + n0 + tx]);
    }
    __syncthreads();
#pragma unroll
    for (int j = 0; j < 4; j++) {
        int n = ty + j * 8;
        dst[(size_t)(n0 + n) * K + k0 + tx] = tile[tx][n];
    }
}

// ---------------- LayerNorm + AdaLN modulate (+ Doob phi) ----------------
__global__ void k_ln_mod(const float* __restrict__ x, const float* __restrict__ lns,
                         const float* __restrict__ lnb, const float* __restrict__ ada,
                         const float* __restrict__ doob_w, const float* __restrict__ doob_b,
                         unsigned short* __restrict__ xbf, float* __restrict__ phi, int has_phi) {
    int row = blockIdx.x;            // b*N + n
    int b = row >> 11;
    int tid = threadIdx.x;
    const float* xr = x + (size_t)row * DD;
    float xv[4];
    float s = 0.f, s2 = 0.f;
#pragma unroll
    for (int j = 0; j < 4; j++) {
        float v = xr[tid + 256 * j];
        xv[j] = v; s += v; s2 += v * v;
    }
    __shared__ float red[8];
    for (int m = 16; m >= 1; m >>= 1) s += __shfl_xor(s, m, 32);
    if ((tid & 31) == 0) red[tid >> 5] = s;
    __syncthreads();
    if (tid == 0) { float tt = 0; for (int i = 0; i < 8; i++) tt += red[i]; red[0] = tt; }
    __syncthreads();
    float mu = red[0] * (1.0f / DD);
    __syncthreads();
    for (int m = 16; m >= 1; m >>= 1) s2 += __shfl_xor(s2, m, 32);
    if ((tid & 31) == 0) red[tid >> 5] = s2;
    __syncthreads();
    if (tid == 0) { float tt = 0; for (int i = 0; i < 8; i++) tt += red[i]; red[0] = tt; }
    __syncthreads();
    float var = red[0] * (1.0f / DD) - mu * mu;
    float rs = rsqrtf(var + 1e-6f);
    float ph = 0.f;
#pragma unroll
    for (int j = 0; j < 4; j++) {
        int d = tid + 256 * j;
        float v = (xv[j] - mu) * rs * lns[d] + lnb[d];
        float mod = v * (1.0f + ada[(size_t)b * HID + 1024 + d]) + ada[(size_t)b * HID + d];
        xbf[(size_t)row * DD + d] = f2bf(mod);
        ph += mod * doob_w[d];
    }
    if (has_phi) {
        for (int m = 16; m >= 1; m >>= 1) ph += __shfl_xor(ph, m, 32);
        __syncthreads();
        if ((tid & 31) == 0) red[tid >> 5] = ph;
        __syncthreads();
        if (tid == 0) { float tt = 0; for (int i = 0; i < 8; i++) tt += red[i]; phi[row] = tt + doob_b[0]; }
    }
}

// ---------------- generic WMMA GEMM (A: MxK bf16, Wt: NoutxK bf16 transposed) ----------------
// mode 0: out bf16 = acc (row-major)
// mode 1: v = acc * exp(ada[b][colOff+col]); optional f32 + bf16 row-major
// mode 2: outf = xin + acc * ada[b][colOff+col] * gamma[col]
// mode 3: bf16 transposed store to (B,H,DH,N)   [token V for flash]
// mode 4: exp-scale then bf16 transposed store to (B,D,N)  [channel qk]
__global__ void k_gemm(const unsigned short* __restrict__ A, const unsigned short* __restrict__ Wt,
                       int K, int Nout, int mode,
                       const float* __restrict__ ada, int colOff,
                       const float* __restrict__ gamma, const float* __restrict__ xin,
                       float* __restrict__ outf, unsigned short* __restrict__ outbf) {
    __shared__ __align__(16) unsigned short As[128 * 32];
    __shared__ __align__(16) unsigned short Bs[64 * 32];
    int tid  = threadIdx.x;
    int lane = tid & 31;
    int wave = tid >> 5;
    int m0 = blockIdx.x * 128;
    int n0 = blockIdx.y * 64;
    v8f acc[4];
#pragma unroll
    for (int j = 0; j < 4; j++) acc[j] = v8f_zero();
    for (int k0 = 0; k0 < K; k0 += 32) {
        __syncthreads();
        {
            int r = tid >> 1, c = (tid & 1) * 16;
            const unsigned short* src = A + (size_t)(m0 + r) * K + k0 + c;
            *(v8bf*)(As + r * 32 + c)     = *(const v8bf*)(src);
            *(v8bf*)(As + r * 32 + c + 8) = *(const v8bf*)(src + 8);
        }
        {
            int r = tid >> 2, c = (tid & 3) * 8;   // 64 rows (n), 32 halves (k)
            *(v8bf*)(Bs + r * 32 + c) = *(const v8bf*)(Wt + (size_t)(n0 + r) * K + k0 + c);
        }
        if (k0 + 32 < K) {
            __builtin_prefetch(A + (size_t)(m0 + (tid >> 1)) * K + k0 + 32, 0, 1);
            __builtin_prefetch(Wt + (size_t)(n0 + (tid >> 2)) * K + k0 + 32, 0, 1);
        }
        __syncthreads();
        v16bf a = frag_a(As + wave * 16 * 32, 32, lane);
#pragma unroll
        for (int j = 0; j < 4; j++) {
            // B frag: col = j*16 + lane%16, 16 contiguous K at (lane/16)*16
            v16bf bb = frag_b_rows(Bs + (size_t)(j * 16 + (lane & 15)) * 32 + ((lane >> 4) << 4));
            acc[j] = wmma_bf16(a, bb, acc[j]);
        }
    }
#pragma unroll
    for (int j = 0; j < 4; j++) {
#pragma unroll
        for (int r = 0; r < 8; r++) {
            int row = m0 + wave * 16 + r + ((lane >> 4) << 3);
            int col = n0 + j * 16 + (lane & 15);
            int b = row >> 11;
            int n = row & 2047;
            size_t idx = (size_t)row * Nout + col;
            float v = acc[j][r];
            if (mode == 0) {
                outbf[idx] = f2bf(v);
            } else if (mode == 1) {
                v *= expf(ada[(size_t)b * HID + colOff + col]);
                if (outf) outf[idx] = v;
                outbf[idx] = f2bf(v);
            } else if (mode == 2) {
                outf[idx] = xin[idx] + v * ada[(size_t)b * HID + colOff + col] * gamma[col];
            } else if (mode == 3) {
                int hh = col >> 6, dl = col & 63;
                outbf[(((size_t)b * HH + hh) * DH + dl) * NN + n] = f2bf(v);
            } else {   // mode 4
                v *= expf(ada[(size_t)b * HID + colOff + col]);
                outbf[((size_t)b * DD + col) * NN + n] = f2bf(v);
            }
        }
    }
}

// ---------------- per-(row, head) sum of squares of qk ----------------
__global__ void k_q2tok(const float* __restrict__ qk, float* __restrict__ q2) {
    int row = blockIdx.x;
    int b = row >> 11;
    int n = row & 2047;
    int tid = threadIdx.x;
    __shared__ float hs[16];
    if (tid < 16) hs[tid] = 0.f;
    __syncthreads();
    const float* r = qk + (size_t)row * DD;
    int d = tid * 4;
    float v0 = r[d], v1 = r[d + 1], v2 = r[d + 2], v3 = r[d + 3];
    atomicAdd(&hs[d >> 6], v0 * v0 + v1 * v1 + v2 * v2 + v3 * v3);
    __syncthreads();
    if (tid < 16) q2[((size_t)b * HH + tid) * NN + n] = hs[tid];
}

// ---------------- flash attention, tied Q=K, Doob tilt ----------------
// qkbf: (B,N,D) bf16; vT: (B,H,DH,N) bf16
__global__ void k_flash(const unsigned short* __restrict__ qkbf, const unsigned short* __restrict__ vT,
                        const float* __restrict__ q2, const float* __restrict__ phi,
                        unsigned short* __restrict__ wout) {
    int bh = blockIdx.x;
    int b = bh / HH, h = bh % HH;
    int wave = threadIdx.x >> 5;
    int lane = threadIdx.x & 31;
    int m0 = blockIdx.y * 128 + wave * 16;
    __shared__ __align__(16) unsigned short Pb[8][16 * 32];
    const unsigned short* Qbase = qkbf + (size_t)b * NN * DD + (size_t)h * DH;
    const unsigned short* Vbase = vT + (size_t)(b * HH + h) * DH * NN;
    v16bf qa0 = frag_a(Qbase + (size_t)m0 * DD, DD, lane);
    v16bf qa1 = frag_a(Qbase + (size_t)m0 * DD + 32, DD, lane);
    float mrow[8], lrow[8];
    v8f o[4];
#pragma unroll
    for (int r = 0; r < 8; r++) { mrow[r] = -3.0e38f; lrow[r] = 0.f; }
#pragma unroll
    for (int j = 0; j < 4; j++) o[j] = v8f_zero();
    unsigned short* Pw = Pb[wave];
    int kh16 = (lane >> 4) << 4;
    for (int n0 = 0; n0 < NN; n0 += 32) {
        v8f s0 = v8f_zero(), s1 = v8f_zero();
        {
            const unsigned short* kr = Qbase + (size_t)(n0 + (lane & 15)) * DD + kh16;
            v16bf b0 = frag_b_rows(kr);
            v16bf b1 = frag_b_rows(kr + 32);
            s0 = wmma_bf16(qa0, b0, s0);
            s0 = wmma_bf16(qa1, b1, s0);
            const unsigned short* kr2 = kr + 16 * DD;
            v16bf c0 = frag_b_rows(kr2);
            v16bf c1 = frag_b_rows(kr2 + 32);
            s1 = wmma_bf16(qa0, c0, s1);
            s1 = wmma_bf16(qa1, c1, s1);
        }
        int nA = n0 + (lane & 15);
        float addA = phi[(size_t)b * NN + nA]      - q2[((size_t)b * HH + h) * NN + nA];
        float addB = phi[(size_t)b * NN + nA + 16] - q2[((size_t)b * HH + h) * NN + nA + 16];
#pragma unroll
        for (int r = 0; r < 8; r++) {
            float sv0 = 2.f * s0[r] + addA;
            float sv1 = 2.f * s1[r] + addB;
            float t = fmaxf(sv0, sv1);
            for (int m = 1; m < 16; m <<= 1) t = fmaxf(t, __shfl_xor(t, m, 32));
            float mnew = fmaxf(mrow[r], t);
            float sc = expf(mrow[r] - mnew);
            float p0 = expf(sv0 - mnew);
            float p1 = expf(sv1 - mnew);
            float ps = p0 + p1;
            for (int m = 1; m < 16; m <<= 1) ps += __shfl_xor(ps, m, 32);
            lrow[r] = lrow[r] * sc + ps;
            mrow[r] = mnew;
#pragma unroll
            for (int j = 0; j < 4; j++) o[j][r] *= sc;
            int prow = r + ((lane >> 4) << 3);
            Pw[prow * 32 + (lane & 15)]      = f2bf(p0);
            Pw[prow * 32 + 16 + (lane & 15)] = f2bf(p1);
        }
        v16bf pa = frag_a(Pw, 32, lane);   // in-wave LDS ops complete in order
#pragma unroll
        for (int j = 0; j < 4; j++) {
            // V^T row = dh column j*16 + lane%16; 16 contiguous keys at n0 + (lane/16)*16
            v16bf vv = frag_b_rows(Vbase + (size_t)(j * 16 + (lane & 15)) * NN + n0 + kh16);
            o[j] = wmma_bf16(pa, vv, o[j]);
        }
    }
#pragma unroll
    for (int r = 0; r < 8; r++) {
        float inv = 1.f / lrow[r];
        int row = m0 + r + ((lane >> 4) << 3);
#pragma unroll
        for (int j = 0; j < 4; j++) {
            wout[(size_t)(b * NN + row) * DD + h * DH + j * 16 + (lane & 15)] =
                f2bf(o[j][r] * inv);
        }
    }
}

// ---------------- q2c: per (b, d) sum over N of qkc^2 ----------------
__global__ void k_q2c(const unsigned short* __restrict__ qkcT, float* __restrict__ q2c) {
    int rowd = blockIdx.x;
    int tid = threadIdx.x;
    const unsigned short* r = qkcT + (size_t)rowd * NN;
    float s = 0.f;
#pragma unroll
    for (int j = 0; j < 8; j++) {
        float v = bf2f(r[tid + 256 * j]);
        s += v * v;
    }
    __shared__ float red[8];
    for (int m = 16; m >= 1; m >>= 1) s += __shfl_xor(s, m, 32);
    if ((tid & 31) == 0) red[tid >> 5] = s;
    __syncthreads();
    if (tid == 0) { float tt = 0; for (int i = 0; i < 8; i++) tt += red[i]; q2c[rowd] = tt; }
}

// ---------------- channel dot: logits = (2 Q Q^T - q2c_c - q2c_d)/sqrt(N)*tau ----------------
__global__ void k_chdot(const unsigned short* __restrict__ qkcT, const float* __restrict__ q2c,
                        const float* __restrict__ tau, float* __restrict__ logits) {
    int bh = blockIdx.x;
    int b = bh / HCC, h = bh % HCC;
    int lane = threadIdx.x & 31;
    int wave = threadIdx.x >> 5;
    int mw = wave * 16;
    const unsigned short* base = qkcT + (size_t)(b * DD + h * DHC) * NN;
    v8f acc[8];
#pragma unroll
    for (int j = 0; j < 8; j++) acc[j] = v8f_zero();
    for (int k0 = 0; k0 < NN; k0 += 32) {
        v16bf a = frag_a(base + (size_t)mw * NN + k0, NN, lane);
#pragma unroll
        for (int j = 0; j < 8; j++) {
            const unsigned short* rp =
                base + (size_t)(j * 16 + (lane & 15)) * NN + k0 + ((lane >> 4) << 4);
            v16bf bb = frag_b_rows(rp);
            acc[j] = wmma_bf16(a, bb, acc[j]);
        }
    }
    float sc = tau[h] * 0.0220970869120796f;   // 1/sqrt(2048)
#pragma unroll
    for (int j = 0; j < 8; j++) {
#pragma unroll
        for (int r = 0; r < 8; r++) {
            int c = mw + r + ((lane >> 4) << 3);
            int d = j * 16 + (lane & 15);
            float v = (2.f * acc[j][r] - q2c[b * DD + h * DHC + c] - q2c[b * DD + h * DHC + d]) * sc;
            logits[((size_t)bh * DHC + c) * DHC + d] = v;
        }
    }
}

// ---------------- softmax over 128 cols ----------------
__global__ void k_softmax128(const float* __restrict__ logits, unsigned short* __restrict__ attnc) {
    int row = blockIdx.x;
    int tid = threadIdx.x;   // 128 threads
    float v = logits[(size_t)row * DHC + tid];
    __shared__ float red[4];
    float m = v;
    for (int s = 16; s >= 1; s >>= 1) m = fmaxf(m, __shfl_xor(m, s, 32));
    if ((tid & 31) == 0) red[tid >> 5] = m;
    __syncthreads();
    if (tid == 0) { float tt = red[0]; for (int i = 1; i < 4; i++) tt = fmaxf(tt, red[i]); red[0] = tt; }
    __syncthreads();
    m = red[0];
    float p = expf(v - m);
    __syncthreads();
    float s = p;
    for (int q = 16; q >= 1; q >>= 1) s += __shfl_xor(s, q, 32);
    if ((tid & 31) == 0) red[tid >> 5] = s;
    __syncthreads();
    if (tid == 0) { float tt = 0; for (int i = 0; i < 4; i++) tt += red[i]; red[0] = tt; }
    __syncthreads();
    attnc[(size_t)row * DHC + tid] = f2bf(p / red[0]);
}

// ---------------- channel PV: chT[b][n][h*128+c] = (attnc @ vc^T)[c][n] ----------------
// vcbf: (B,N,D) bf16 row-major -> rows of it are exactly the B-fragment lanes.
__global__ void k_chwc(const unsigned short* __restrict__ attnc, const unsigned short* __restrict__ vcbf,
                       unsigned short* __restrict__ chT) {
    int bh = blockIdx.x;
    int b = bh / HCC, h = bh % HCC;
    int n0 = blockIdx.y * 64;
    int lane = threadIdx.x & 31;
    int wave = threadIdx.x >> 5;
    int mw = wave * 16;
    const unsigned short* Ab = attnc + (size_t)bh * DHC * DHC;
    v8f acc[4];
#pragma unroll
    for (int j = 0; j < 4; j++) acc[j] = v8f_zero();
    for (int k0 = 0; k0 < DHC; k0 += 32) {
        v16bf a = frag_a(Ab + (size_t)mw * DHC + k0, DHC, lane);
#pragma unroll
        for (int j = 0; j < 4; j++) {
            int n = n0 + j * 16 + (lane & 15);
            const unsigned short* rp =
                vcbf + (size_t)(b * NN + n) * DD + h * DHC + k0 + ((lane >> 4) << 4);
            v16bf vv = frag_b_rows(rp);
            acc[j] = wmma_bf16(a, vv, acc[j]);
        }
    }
#pragma unroll
    for (int j = 0; j < 4; j++) {
#pragma unroll
        for (int r = 0; r < 8; r++) {
            int c = mw + r + ((lane >> 4) << 3);
            int n = n0 + j * 16 + (lane & 15);
            chT[(size_t)(b * NN + n) * DD + h * DHC + c] = f2bf(acc[j][r]);
        }
    }
}

extern "C" void kernel_launch(void* const* d_in, const int* in_sizes, int n_in,
                              void* d_out, int out_size, void* d_ws, size_t ws_size,
                              hipStream_t stream) {
    (void)in_sizes; (void)n_in; (void)out_size; (void)ws_size;
    const float* x      = (const float*)d_in[0];
    const float* t      = (const float*)d_in[1];
    const float* n1s    = (const float*)d_in[2];
    const float* n1b    = (const float*)d_in[3];
    const float* n2s    = (const float*)d_in[4];
    const float* n2b    = (const float*)d_in[5];
    const float* tc_w1  = (const float*)d_in[6];
    const float* tc_b1  = (const float*)d_in[7];
    const float* tc_w2  = (const float*)d_in[8];
    const float* tc_b2  = (const float*)d_in[9];
    const float* tc_wa  = (const float*)d_in[10];
    const float* tc_ba  = (const float*)d_in[11];
    const float* tc_wc  = (const float*)d_in[12];
    const float* tc_bc  = (const float*)d_in[13];
    const float* qk_w   = (const float*)d_in[14];
    const float* v_w    = (const float*)d_in[15];
    const float* out_w  = (const float*)d_in[16];
    const float* doob_w = (const float*)d_in[17];
    const float* doob_b = (const float*)d_in[18];
    const float* cqk_w  = (const float*)d_in[19];
    const float* cv_w   = (const float*)d_in[20];
    const float* cout_w = (const float*)d_in[21];
    const float* tau    = (const float*)d_in[22];
    const float* g1     = (const float*)d_in[23];
    const float* g2     = (const float*)d_in[24];
    float* out = (float*)d_out;

    char* ws = (char*)d_ws;
    size_t off = 0;
    auto alloc = [&](size_t bytes) -> char* {
        char* p = ws + off;
        off += (bytes + 255) & ~(size_t)255;
        return p;
    };
    const size_t ND  = (size_t)BB * NN * DD;   // 4194304
    const size_t WEL = (size_t)DD * DD;        // 1048576

    float* emb  = (float*)alloc((size_t)BB * HID * 4);
    float* h1   = (float*)alloc((size_t)BB * HID * 4);
    float* h2   = (float*)alloc((size_t)BB * HID * 4);
    float* adaA = (float*)alloc((size_t)BB * HID * 4);
    float* adaC = (float*)alloc((size_t)BB * HID * 4);
    unsigned short* wqk_t   = (unsigned short*)alloc(WEL * 2);
    unsigned short* wv_t    = (unsigned short*)alloc(WEL * 2);
    unsigned short* wout_t  = (unsigned short*)alloc(WEL * 2);
    unsigned short* wcqk_t  = (unsigned short*)alloc(WEL * 2);
    unsigned short* wcv_t   = (unsigned short*)alloc(WEL * 2);
    unsigned short* wcout_t = (unsigned short*)alloc(WEL * 2);
    unsigned short* x1bf  = (unsigned short*)alloc(ND * 2);
    float* phi  = (float*)alloc((size_t)BB * NN * 4);
    float* qkf  = (float*)alloc(ND * 4);
    unsigned short* qkbf  = (unsigned short*)alloc(ND * 2);
    unsigned short* vTbf  = (unsigned short*)alloc(ND * 2);   // (B,H,DH,N)
    float* q2   = (float*)alloc((size_t)BB * HH * NN * 4);
    unsigned short* wattn = (unsigned short*)alloc(ND * 2);
    float* xmid = (float*)alloc(ND * 4);
    unsigned short* x2bf  = (unsigned short*)alloc(ND * 2);
    unsigned short* qkcT  = (unsigned short*)alloc(ND * 2);   // (B,D,N)
    unsigned short* vcbf  = (unsigned short*)alloc(ND * 2);   // (B,N,D)
    float* q2c  = (float*)alloc((size_t)BB * DD * 4);
    float* logc = (float*)alloc((size_t)BB * HCC * DHC * DHC * 4);
    unsigned short* attnc = (unsigned short*)alloc((size_t)BB * HCC * DHC * DHC * 2);
    unsigned short* chT   = (unsigned short*)alloc(ND * 2);

    // ---- time conditioning ----
    k_embed<<<(BB * HID) / 256, 256, 0, stream>>>(t, emb);
    k_gemv<<<HID / 256, 256, 0, stream>>>(emb, tc_w1, tc_b1, h1, HID, HID, 1);
    k_gemv<<<HID / 256, 256, 0, stream>>>(h1, tc_w2, tc_b2, h2, HID, HID, 1);
    k_gemv<<<HID / 256, 256, 0, stream>>>(h2, tc_wa, tc_ba, adaA, HID, HID, 0);
    k_gemv<<<HID / 256, 256, 0, stream>>>(h2, tc_wc, tc_bc, adaC, HID, HID, 0);

    // ---- weight conversion to transposed bf16 ----
    dim3 wtGrid(DD / 32, DD / 32);
    dim3 wtBlk(32, 8);
    k_f2bf_t<<<wtGrid, wtBlk, 0, stream>>>(qk_w, wqk_t, DD, DD);
    k_f2bf_t<<<wtGrid, wtBlk, 0, stream>>>(v_w, wv_t, DD, DD);
    k_f2bf_t<<<wtGrid, wtBlk, 0, stream>>>(out_w, wout_t, DD, DD);
    k_f2bf_t<<<wtGrid, wtBlk, 0, stream>>>(cqk_w, wcqk_t, DD, DD);
    k_f2bf_t<<<wtGrid, wtBlk, 0, stream>>>(cv_w, wcv_t, DD, DD);
    k_f2bf_t<<<wtGrid, wtBlk, 0, stream>>>(cout_w, wcout_t, DD, DD);

    dim3 gemmGrid((BB * NN) / 128, DD / 64);

    // ---- token attention path ----
    k_ln_mod<<<BB * NN, 256, 0, stream>>>(x, n1s, n1b, adaA, doob_w, doob_b, x1bf, phi, 1);
    k_gemm<<<gemmGrid, 256, 0, stream>>>(x1bf, wqk_t, DD, DD, 1, adaA, 3072,
                                         nullptr, nullptr, qkf, qkbf);
    k_gemm<<<gemmGrid, 256, 0, stream>>>(x1bf, wv_t, DD, DD, 3, nullptr, 0,
                                         nullptr, nullptr, nullptr, vTbf);
    k_q2tok<<<BB * NN, 256, 0, stream>>>(qkf, q2);
    k_flash<<<dim3(BB * HH, NN / 128), 256, 0, stream>>>(qkbf, vTbf, q2, phi, wattn);
    k_gemm<<<gemmGrid, 256, 0, stream>>>(wattn, wout_t, DD, DD, 2, adaA, 2048,
                                         g1, x, xmid, nullptr);

    // ---- channel attention path ----
    k_ln_mod<<<BB * NN, 256, 0, stream>>>(xmid, n2s, n2b, adaC, doob_w, doob_b, x2bf, nullptr, 0);
    k_gemm<<<gemmGrid, 256, 0, stream>>>(x2bf, wcqk_t, DD, DD, 4, adaC, 3072,
                                         nullptr, nullptr, nullptr, qkcT);
    k_gemm<<<gemmGrid, 256, 0, stream>>>(x2bf, wcv_t, DD, DD, 0, nullptr, 0,
                                         nullptr, nullptr, nullptr, vcbf);
    k_q2c<<<BB * DD, 256, 0, stream>>>(qkcT, q2c);
    k_chdot<<<BB * HCC, 256, 0, stream>>>(qkcT, q2c, tau, logc);
    k_softmax128<<<BB * HCC * DHC, 128, 0, stream>>>(logc, attnc);
    k_chwc<<<dim3(BB * HCC, NN / 64), 256, 0, stream>>>(attnc, vcbf, chT);
    k_gemm<<<gemmGrid, 256, 0, stream>>>(chT, wcout_t, DD, DD, 2, adaC, 2048,
                                         g2, xmid, out, nullptr);
}
